// NGNN_GCNConv_26877905339081
// MI455X (gfx1250) — matrix operand order repaired and checked by reading, not compiled, not run on hardware
//
#include <hip/hip_runtime.h>
#include <hip/hip_bf16.h>

// ---------------------------------------------------------------------------
// NGNN GCNConv forward for MI455X (gfx1250, wave32, WMMA).
// Pipeline:
//   1) zero scratch (deg_out, deg_in, agg)
//   2) degree accumulation (fp32 atomics)
//   3) in-place rsqrt(max(deg,1)) -> norm_src / norm_dst
//   4) weight transpose + fp32->bf16 convert (WMMA B-fragment layout)
//   5) SpMM: agg[dst] += x[src] * norm_src[src]   (L2-resident atomics)
//   6) GEMM1: (agg * norm_dst) @ W_conv + b, ReLU -> h1 (bf16)   [WMMA bf16]
//   7) GEMM2: h1 @ W_fc + b, ReLU -> h2 (bf16)                   [WMMA bf16]
//   8) GEMM3: h2 @ W_fc2 + b -> d_out (fp32)                     [WMMA bf16]
// GEMM blocking: 64 rows x 256 cols per block (4 waves); A K-slice staged in
// LDS (bank-conflict-free, shared by all waves), B slabs stream from L2.
// ---------------------------------------------------------------------------

typedef __bf16 bf16_t;
typedef __attribute__((ext_vector_type(16))) __bf16 v16bf;
typedef __attribute__((ext_vector_type(8)))  __bf16 v8bf;
typedef __attribute__((ext_vector_type(8)))  float  v8f;
typedef __attribute__((ext_vector_type(4)))  float  v4f;

static constexpr int F_IN  = 128;
static constexpr int F_H   = 256;
static constexpr int F_OUT = 256;

static constexpr int LDS_STRIDE = 40;   // 32 k + 8 pad (bf16) -> 80B row stride

// ---------------------------------------------------------------- zero fill
__global__ void k_zero_f32(float* __restrict__ p, long n) {
    long i = (long)blockIdx.x * blockDim.x + threadIdx.x;
    if (i < n) p[i] = 0.0f;
}

// ------------------------------------------------------------------ degrees
__global__ void k_degrees(const int* __restrict__ src, const int* __restrict__ dst,
                          float* __restrict__ deg_out, float* __restrict__ deg_in,
                          int E) {
    int e = blockIdx.x * blockDim.x + threadIdx.x;
    if (e < E) {
        atomicAdd(&deg_out[src[e]], 1.0f);
        atomicAdd(&deg_in [dst[e]], 1.0f);
    }
}

// --------------------------------------------------- deg -> rsqrt(max(d,1))
__global__ void k_norm_inplace(float* __restrict__ p, int n) {
    int i = blockIdx.x * blockDim.x + threadIdx.x;
    if (i < n) p[i] = __frsqrt_rn(fmaxf(p[i], 1.0f));
}

// ------------------------- W [K][NO] fp32 -> Wt [NO][K] bf16 (B-fragment)
__global__ void k_prep_wT(const float* __restrict__ W, bf16_t* __restrict__ Wt,
                          int K, int NO) {
    int idx = blockIdx.x * blockDim.x + threadIdx.x;
    if (idx >= K * NO) return;
    int n = idx / K;
    int k = idx - n * K;
    Wt[idx] = (bf16_t)W[(long)k * NO + n];
}

// --------------------------------------------------------------------- SpMM
// one edge per wave32; lane owns a float4 feature chunk (128 feats = 32*4)
__global__ void k_spmm_atomic(const float* __restrict__ x,
                              const int* __restrict__ src,
                              const int* __restrict__ dst,
                              const float* __restrict__ norm_src,
                              float* __restrict__ agg, int E) {
    long tid = (long)blockIdx.x * blockDim.x + threadIdx.x;
    int  e   = (int)(tid >> 5);
    int  c   = (int)(tid & 31) << 2;            // feature offset 0..124
    if (e >= E) return;
    int   s = src[e];
    int   d = dst[e];
    float w = norm_src[s];
    v4f xv = *(const v4f*)(x + (long)s * F_IN + c);
    float* out = agg + (long)d * F_IN + c;
    atomicAdd(out + 0, xv.x * w);
    atomicAdd(out + 1, xv.y * w);
    atomicAdd(out + 2, xv.z * w);
    atomicAdd(out + 3, xv.w * w);
}

// -------------------------------------------------------------- WMMA GEMM
// block = 128 threads (4 waves). Block owns 64 rows x 256 cols.
// Wave w owns output columns [w*64, w*64+64): 4 m-subtiles x 4 n-tiles.
// A K-slice (64 rows x 32 k, bf16) is staged in LDS each K-step by all 128
// threads, then each wave reads its fragments with ds_load_b128 (padded
// stride -> conflict-free). B fragments stream from global/L2 (Wt is [n][k],
// so a B fragment is one contiguous 32B chunk per lane).
// Fragment layouts used here:
//   A (16x32): lane holds row (lane&15); contiguous K chunk at half*16.
//   B (32x16): lane holds col (lane&15); contiguous K chunk at half*16.
//   C/D:       lane holds col (lane&15); rows r + half*8, r=0..7.
template <int K, bool A_F32, bool RELU, bool OUT_BF16>
__global__ __launch_bounds__(128)
void k_gemm_wmma(const void* __restrict__ Ain,
                 const float* __restrict__ norm,     // used iff A_F32
                 const bf16_t* __restrict__ Wt,      // [256][K] bf16
                 const float* __restrict__ bias,     // [256]
                 void* __restrict__ Out,             // [nRows][256]
                 int nRows) {
    __shared__ bf16_t As[64 * LDS_STRIDE];

    const int lane    = threadIdx.x & 31;
    const int wave    = threadIdx.x >> 5;    // 0..3
    const int half    = lane >> 4;           // 0/1
    const int mr      = lane & 15;
    const int mbase   = blockIdx.x * 64;
    const int colbase = wave * 64;

    v8f acc[4][4] = {};

#pragma unroll
    for (int kk = 0; kk < K; kk += 32) {
        // ---- stage A K-slice into LDS: 256 chunks of 8 bf16, 2 per thread
#pragma unroll
        for (int cc = 0; cc < 2; ++cc) {
            const int c  = threadIdx.x * 2 + cc;
            const int r  = c >> 2;             // 0..63
            const int ko = (c & 3) * 8;        // 0,8,16,24
            int grow = mbase + r;
            if (grow > nRows - 1) grow = nRows - 1;   // branchless clamp
            if constexpr (A_F32) {
                const float* Af  = (const float*)Ain + (long)grow * K + kk + ko;
                const float  scl = norm[grow];
                const v4f f0 = *(const v4f*)(Af);
                const v4f f1 = *(const v4f*)(Af + 4);
                v8bf t;
#pragma unroll
                for (int i = 0; i < 4; ++i) {
                    t[i]     = (bf16_t)(f0[i] * scl);
                    t[4 + i] = (bf16_t)(f1[i] * scl);
                }
                *(v8bf*)&As[r * LDS_STRIDE + ko] = t;
            } else {
                const bf16_t* Ab = (const bf16_t*)Ain + (long)grow * K + kk + ko;
                *(v8bf*)&As[r * LDS_STRIDE + ko] = *(const v8bf*)Ab;
            }
        }
        __syncthreads();

        // ---- load A fragments for the 4 m-subtiles (conflict-free ds reads)
        v16bf a[4];
#pragma unroll
        for (int mt = 0; mt < 4; ++mt) {
            const bf16_t* rp = &As[(mt * 16 + mr) * LDS_STRIDE];
            const v8bf a0 = *(const v8bf*)(rp + half * 16);
            const v8bf a1 = *(const v8bf*)(rp + half * 16 + 8);
#pragma unroll
            for (int i = 0; i < 8; ++i) { a[mt][i] = a0[i]; a[mt][8 + i] = a1[i]; }
        }

        // ---- 16 WMMAs: B loaded once per n-tile, reused across 4 m-subtiles
#pragma unroll
        for (int nt = 0; nt < 4; ++nt) {
            const bf16_t* Bp = Wt + (long)(colbase + nt * 16 + mr) * K + kk + half * 16;
            const v16bf b = *(const v16bf*)Bp;
#pragma unroll
            for (int mt = 0; mt < 4; ++mt) {
                acc[mt][nt] = __builtin_amdgcn_wmma_f32_16x16x32_bf16(
                    false, a[mt], false, b, (short)0, acc[mt][nt], false, false);
            }
        }
        __syncthreads();   // protect LDS before next staging (WAR)
    }

    // ---- epilogue: bias (+ReLU) and store
#pragma unroll
    for (int nt = 0; nt < 4; ++nt) {
        const int   col = colbase + nt * 16 + mr;
        const float bv  = bias[col];
#pragma unroll
        for (int mt = 0; mt < 4; ++mt) {
#pragma unroll
            for (int r = 0; r < 8; ++r) {
                const int orow = mbase + mt * 16 + r + half * 8;
                if (orow < nRows) {
                    float v = acc[mt][nt][r] + bv;
                    if constexpr (RELU) v = fmaxf(v, 0.0f);
                    if constexpr (OUT_BF16)
                        ((bf16_t*)Out)[(long)orow * 256 + col] = (bf16_t)v;
                    else
                        ((float*)Out)[(long)orow * 256 + col] = v;
                }
            }
        }
    }
}

// ---------------------------------------------------------------------------
extern "C" void kernel_launch(void* const* d_in, const int* in_sizes, int n_in,
                              void* d_out, int out_size, void* d_ws, size_t ws_size,
                              hipStream_t stream) {
    const float* x      = (const float*)d_in[0];
    const int*   src    = (const int*)  d_in[1];
    const int*   dst    = (const int*)  d_in[2];
    const float* W_conv = (const float*)d_in[3];
    const float* b_conv = (const float*)d_in[4];
    const float* W_fc   = (const float*)d_in[5];
    const float* b_fc   = (const float*)d_in[6];
    const float* W_fc2  = (const float*)d_in[7];
    const float* b_fc2  = (const float*)d_in[8];

    const int N = in_sizes[0] / F_IN;   // 50000
    const int E = in_sizes[1];          // 600000

    // -------- workspace layout (zeroed region first, agg aliased with h2)
    char*  ws  = (char*)d_ws;
    size_t off = 0;
    float* deg_out = (float*)(ws + off); off += (size_t)N * 4;
    float* deg_in  = (float*)(ws + off); off += (size_t)N * 4;
    float* agg     = (float*)(ws + off);                    // N*128 fp32
    bf16_t* h2     = (bf16_t*)agg;                          // alias: N*256 bf16
    off += (size_t)N * F_IN * 4;
    bf16_t* Wt_conv = (bf16_t*)(ws + off); off += (size_t)F_IN * F_H  * 2;
    bf16_t* Wt_fc   = (bf16_t*)(ws + off); off += (size_t)F_H  * F_H  * 2;
    bf16_t* Wt_fc2  = (bf16_t*)(ws + off); off += (size_t)F_H  * F_OUT * 2;
    bf16_t* h1      = (bf16_t*)(ws + off); off += (size_t)N * F_H * 2;

    // 1) zero deg_out|deg_in|agg (contiguous)
    {
        long nz = (long)2 * N + (long)N * F_IN;
        k_zero_f32<<<dim3((unsigned)((nz + 255) / 256)), dim3(256), 0, stream>>>(deg_out, nz);
    }

    // 2) degrees
    k_degrees<<<dim3((E + 255) / 256), dim3(256), 0, stream>>>(src, dst, deg_out, deg_in, E);

    // 3) norms in place (both arrays contiguous)
    k_norm_inplace<<<dim3((2 * N + 255) / 256), dim3(256), 0, stream>>>(deg_out, 2 * N);

    // 4) weight prep (transpose + bf16)
    k_prep_wT<<<dim3((F_IN * F_H  + 255) / 256), dim3(256), 0, stream>>>(W_conv, Wt_conv, F_IN, F_H);
    k_prep_wT<<<dim3((F_H  * F_H  + 255) / 256), dim3(256), 0, stream>>>(W_fc,   Wt_fc,   F_H,  F_H);
    k_prep_wT<<<dim3((F_H  * F_OUT + 255) / 256), dim3(256), 0, stream>>>(W_fc2,  Wt_fc2,  F_H,  F_OUT);

    // 5) SpMM gather/scatter (one edge per wave32)
    {
        long tot = (long)E * 32;
        k_spmm_atomic<<<dim3((unsigned)((tot + 255) / 256)), dim3(256), 0, stream>>>(
            x, src, dst, /*norm_src=*/deg_out, agg, E);
    }

    // 6-8) fused GEMM chain: 64-row M tiles per block
    const int Mt = (N + 63) / 64;
    k_gemm_wmma<F_IN, true,  true,  true ><<<dim3(Mt), dim3(128), 0, stream>>>(
        agg, /*norm_dst=*/deg_in, Wt_conv, b_conv, h1, N);
    k_gemm_wmma<F_H,  false, true,  true ><<<dim3(Mt), dim3(128), 0, stream>>>(
        h1, nullptr, Wt_fc, b_fc, h2, N);
    k_gemm_wmma<F_H,  false, false, false><<<dim3(Mt), dim3(128), 0, stream>>>(
        h2, nullptr, Wt_fc2, b_fc2, d_out, N);
}
